// TransformerLayer_85925115724470
// MI455X (gfx1250) — matrix operand order repaired and checked by reading, not compiled, not run on hardware
//
#include <hip/hip_runtime.h>
#include <hip/hip_bf16.h>
#include <math.h>

// ---------------------------------------------------------------------------
// Transformer encoder layer for MI455X (gfx1250), bf16 WMMA everywhere.
// B=8, S=1024, D=1024, H=16, DH=64, FF=4096.
// GEMMs: 256x64 macro-tile, 8 waves, each wave 32x64 (8 WMMA / K-stage),
// A-tile staged with CDNA5 async global->LDS DMA (ASYNCcnt).
// ---------------------------------------------------------------------------

typedef unsigned short u16;
typedef __attribute__((ext_vector_type(16))) __bf16 bf16x16;
typedef __attribute__((ext_vector_type(8)))  float  f32x8;

union FragB { bf16x16 v; u16 u[16]; uint4 q[2]; };

#define DEV __device__ __forceinline__

constexpr int Bc   = 8;
constexpr int Sq   = 1024;
constexpr int Dm   = 1024;
constexpr int Hh   = 16;
constexpr int DHd  = 64;
constexpr int DFF  = 4 * Dm;
constexpr int Mrows = Bc * Sq;          // 8192

DEV u16 f2bf(float f) {
  unsigned u = __float_as_uint(f);
  unsigned r = u + 0x7FFFu + ((u >> 16) & 1u);   // round to nearest even
  return (u16)(r >> 16);
}

// CDNA5 async DMA: memory -> LDS without VGPR round trip (tracked by ASYNCcnt).
// Generic->LDS address mapping truncates to addr[31:0] (ISA 10.2), so the low
// 32 bits of a generic pointer to __shared__ are a valid LDS byte offset.
DEV void async_load_b128(void* lds_ptr, const void* gptr) {
  unsigned lds_off = (unsigned)(uintptr_t)lds_ptr;
  unsigned long long ga = (unsigned long long)(uintptr_t)gptr;
  asm volatile("global_load_async_to_lds_b128 %0, %1, off"
               :: "v"(lds_off), "v"(ga)
               : "memory");
}
DEV void wait_async0() {
  asm volatile("s_wait_asynccnt 0x0" ::: "memory");
}

// ---------------------------------------------------------------------------
// fp32 -> bf16 elementwise conversion (grid-stride)
// ---------------------------------------------------------------------------
__global__ __launch_bounds__(256)
void cvt_bf16_k(const float* __restrict__ in, u16* __restrict__ out, size_t n) {
  size_t i = (size_t)blockIdx.x * blockDim.x + threadIdx.x;
  size_t stride = (size_t)gridDim.x * blockDim.x;
  for (; i < n; i += stride) out[i] = f2bf(in[i]);
}

// ---------------------------------------------------------------------------
// Generic 256x64-tile bf16 WMMA GEMM:  C[M,N] = A[M,K] @ B[K,N]  (+epilogue)
//   EPI 0: C f32
//   EPI 1: C f32 = acc + bias[n] + resid[m,n]
//   EPI 2: C bf16 = gelu_exact(acc + bias[n])
// ---------------------------------------------------------------------------
template<int EPI>
__global__ __launch_bounds__(256)
void gemm_bf16_k(const u16* __restrict__ A, const u16* __restrict__ Bm,
                 const float* __restrict__ bias, const float* __restrict__ resid,
                 float* __restrict__ Cf, u16* __restrict__ Cb,
                 int Msz, int Nsz, int Ksz) {
  __shared__ u16 As[256][40];   // 256 x 32 K-slab, padded
  __shared__ u16 Bt[64][40];    // transposed: [n][k]
  (void)Msz;

  const int tid  = threadIdx.x;
  const int wv   = tid >> 5;
  const int lane = tid & 31;
  const int col  = lane & 15;
  const int half = lane >> 4;
  const int rbase = half * 8;
  const int n0 = blockIdx.x * 64;
  const int m0 = blockIdx.y * 256;
  const int wm = wv * 32;

  f32x8 acc[2][4];
#pragma unroll
  for (int s = 0; s < 2; ++s)
#pragma unroll
    for (int t = 0; t < 4; ++t) acc[s][t] = (f32x8)0.0f;

  for (int k0 = 0; k0 < Ksz; k0 += 32) {
    // stage A (256x32) via async global->LDS DMA: 1024 x b128, 4 per thread
#pragma unroll
    for (int j = 0; j < 4; ++j) {
      int idx = tid + j * 256;
      int r = idx >> 2, c8 = (idx & 3) * 8;
      async_load_b128(&As[r][c8], A + (size_t)(m0 + r) * Ksz + k0 + c8);
    }
    // stage B transposed (32x64 -> Bt[64][32])
    {
      int kr = tid >> 3, n8 = (tid & 7) * 8;
      union { uint4 q; u16 s[8]; } t;
      t.q = *(const uint4*)(Bm + (size_t)(k0 + kr) * Nsz + n0 + n8);
#pragma unroll
      for (int j = 0; j < 8; ++j) Bt[n8 + j][kr] = t.s[j];
    }
    if (k0 + 32 < Ksz)
      __builtin_prefetch(Bm + (size_t)(k0 + 32) * Nsz + n0, 0, 1);
    wait_async0();
    __syncthreads();

    FragB af[2];
#pragma unroll
    for (int s = 0; s < 2; ++s) {
      const u16* ar = &As[wm + s * 16 + col][0];
      int kb = half ? 8 : 0;
      af[s].q[0] = *(const uint4*)(ar + kb);
      af[s].q[1] = *(const uint4*)(ar + kb + 16);
    }
    FragB bfg[4];
#pragma unroll
    for (int tn = 0; tn < 4; ++tn) {
      const u16* br = &Bt[tn * 16 + col][0];
      int kb = half ? 8 : 0;
      bfg[tn].q[0] = *(const uint4*)(br + kb);
      bfg[tn].q[1] = *(const uint4*)(br + kb + 16);
    }
#pragma unroll
    for (int s = 0; s < 2; ++s)
#pragma unroll
      for (int tn = 0; tn < 4; ++tn)
        acc[s][tn] = __builtin_amdgcn_wmma_f32_16x16x32_bf16(
            false, af[s].v, false, bfg[tn].v, (short)0, acc[s][tn], false, false);
    __syncthreads();
  }

  // epilogue
#pragma unroll
  for (int s = 0; s < 2; ++s) {
#pragma unroll
    for (int tn = 0; tn < 4; ++tn) {
      int n = n0 + tn * 16 + col;
      float bv = (EPI >= 1) ? bias[n] : 0.0f;
#pragma unroll
      for (int i = 0; i < 8; ++i) {
        int m = m0 + wm + s * 16 + rbase + i;
        float v = acc[s][tn][i] + bv;
        if (EPI == 1) {
          v += resid[(size_t)m * Nsz + n];
          Cf[(size_t)m * Nsz + n] = v;
        } else if (EPI == 2) {
          float g = 0.5f * v * (1.0f + erff(v * 0.70710678118654752f));
          Cb[(size_t)m * Nsz + n] = f2bf(g);
        } else {
          Cf[(size_t)m * Nsz + n] = v;
        }
      }
    }
  }
}

// ---------------------------------------------------------------------------
// QKV projection: per (m-tile, head, {q,k,v}) compute X[8192,1024] @ W[h][1024,64]
// Output bf16 in [B,H,S,DH]. Same 256x64 tiling + async A staging.
// ---------------------------------------------------------------------------
__global__ __launch_bounds__(256)
void qkv_proj_k(const u16* __restrict__ X,
                const u16* __restrict__ Wqb, const u16* __restrict__ Wkb,
                const u16* __restrict__ Wvb,
                u16* __restrict__ Qo, u16* __restrict__ Ko, u16* __restrict__ Vo) {
  __shared__ u16 As[256][40];
  __shared__ u16 Bt[64][40];

  const int tid  = threadIdx.x;
  const int wv   = tid >> 5;
  const int lane = tid & 31;
  const int col  = lane & 15;
  const int half = lane >> 4;
  const int rbase = half * 8;
  const int m0 = blockIdx.x * 256;
  const int h  = blockIdx.y;
  const int which = blockIdx.z;
  const u16* Bm = (which == 0 ? Wqb : which == 1 ? Wkb : Wvb) + (size_t)h * Dm * DHd;
  u16* Out = which == 0 ? Qo : which == 1 ? Ko : Vo;
  const int wm = wv * 32;

  f32x8 acc[2][4];
#pragma unroll
  for (int s = 0; s < 2; ++s)
#pragma unroll
    for (int t = 0; t < 4; ++t) acc[s][t] = (f32x8)0.0f;

  for (int k0 = 0; k0 < Dm; k0 += 32) {
#pragma unroll
    for (int j = 0; j < 4; ++j) {
      int idx = tid + j * 256;
      int r = idx >> 2, c8 = (idx & 3) * 8;
      async_load_b128(&As[r][c8], X + (size_t)(m0 + r) * Dm + k0 + c8);
    }
    {
      int kr = tid >> 3, n8 = (tid & 7) * 8;
      union { uint4 q; u16 s[8]; } t;
      t.q = *(const uint4*)(Bm + (size_t)(k0 + kr) * DHd + n8);
#pragma unroll
      for (int j = 0; j < 8; ++j) Bt[n8 + j][kr] = t.s[j];
    }
    if (k0 + 32 < Dm)
      __builtin_prefetch(Bm + (size_t)(k0 + 32) * DHd, 0, 1);
    wait_async0();
    __syncthreads();

    FragB af[2];
#pragma unroll
    for (int s = 0; s < 2; ++s) {
      const u16* ar = &As[wm + s * 16 + col][0];
      int kb = half ? 8 : 0;
      af[s].q[0] = *(const uint4*)(ar + kb);
      af[s].q[1] = *(const uint4*)(ar + kb + 16);
    }
    FragB bfg[4];
#pragma unroll
    for (int tn = 0; tn < 4; ++tn) {
      const u16* br = &Bt[tn * 16 + col][0];
      int kb = half ? 8 : 0;
      bfg[tn].q[0] = *(const uint4*)(br + kb);
      bfg[tn].q[1] = *(const uint4*)(br + kb + 16);
    }
#pragma unroll
    for (int s = 0; s < 2; ++s)
#pragma unroll
      for (int tn = 0; tn < 4; ++tn)
        acc[s][tn] = __builtin_amdgcn_wmma_f32_16x16x32_bf16(
            false, af[s].v, false, bfg[tn].v, (short)0, acc[s][tn], false, false);
    __syncthreads();
  }

#pragma unroll
  for (int s = 0; s < 2; ++s) {
#pragma unroll
    for (int tn = 0; tn < 4; ++tn) {
      int n = tn * 16 + col;
#pragma unroll
      for (int i = 0; i < 8; ++i) {
        int m = m0 + wm + s * 16 + rbase + i;
        int srow = m & (Sq - 1);
        int bidx = m >> 10;
        Out[(((size_t)bidx * Hh + h) * Sq + srow) * DHd + n] = f2bf(acc[s][tn][i]);
      }
    }
  }
}

// ---------------------------------------------------------------------------
// Flash attention: one workgroup per (q-block of 128 rows, head, batch).
// 8 waves x 16 query rows. Iterates key blocks of 64 with online softmax.
// Q.K^T: K rows feed WMMA B-fragments directly (B-frag == row-major K).
// P.V  : P via LDS round-trip (A-frag), V transpose-staged in LDS (B-frag).
// Output: bf16 concat [B,S,H*DH].
// ---------------------------------------------------------------------------
__global__ __launch_bounds__(256)
void flash_attn_k(const u16* __restrict__ Q, const u16* __restrict__ Kg,
                  const u16* __restrict__ V, u16* __restrict__ Ctx) {
  __shared__ u16 Vt[DHd][72];      // [e][key], transposed V block
  __shared__ u16 Pl[8][16][72];    // per-wave P tiles (16 q-rows x 64 keys)

  const int tid  = threadIdx.x;
  const int wv   = tid >> 5;
  const int lane = tid & 31;
  const int col  = lane & 15;
  const int half = lane >> 4;
  const int rbase = half * 8;
  const int b = blockIdx.z, h = blockIdx.y;
  const int q0 = blockIdx.x * 128 + wv * 16;
  const float scale = 0.125f;      // 1/sqrt(64)

  const u16* qbase = Q + (((size_t)b * Hh + h) * Sq) * DHd;
  const u16* kbase = Kg + (((size_t)b * Hh + h) * Sq) * DHd;
  const u16* vbase = V + (((size_t)b * Hh + h) * Sq) * DHd;

  // Q fragments for this wave's 16 rows (k-steps 0..31, 32..63)
  FragB qa[2];
  {
    const u16* qrow = qbase + (size_t)(q0 + col) * DHd;
#pragma unroll
    for (int s = 0; s < 2; ++s) {
      int kb = s * 32 + (half ? 8 : 0);
      qa[s].q[0] = *(const uint4*)(qrow + kb);
      qa[s].q[1] = *(const uint4*)(qrow + kb + 16);
    }
  }

  f32x8 o[4];
#pragma unroll
  for (int t = 0; t < 4; ++t) o[t] = (f32x8)0.0f;
  float runmax[8], Lsum[8];
#pragma unroll
  for (int i = 0; i < 8; ++i) { runmax[i] = -3.0e38f; Lsum[i] = 0.0f; }

  for (int k0 = 0; k0 < Sq; k0 += 64) {
    // transpose-stage V block: Vt[e][key]
    {
      int key = tid >> 2;
      int e0 = (tid & 3) * 16;
      const u16* vrow = vbase + (size_t)(k0 + key) * DHd + e0;
      union { uint4 q; u16 s[8]; } a0, a1;
      a0.q = *(const uint4*)(vrow);
      a1.q = *(const uint4*)(vrow + 8);
#pragma unroll
      for (int j = 0; j < 8; ++j) {
        Vt[e0 + j][key]     = a0.s[j];
        Vt[e0 + 8 + j][key] = a1.s[j];
      }
    }
    __syncthreads();

    // scores = Q (16x64) . K^T (64 x 64 keys), f32 accum
    f32x8 sacc[4];
#pragma unroll
    for (int t = 0; t < 4; ++t) sacc[t] = (f32x8)0.0f;
#pragma unroll
    for (int tn = 0; tn < 4; ++tn) {
      const u16* krow = kbase + (size_t)(k0 + tn * 16 + col) * DHd;
#pragma unroll
      for (int s = 0; s < 2; ++s) {
        FragB kf;
        int kb = s * 32 + (half ? 8 : 0);
        kf.q[0] = *(const uint4*)(krow + kb);
        kf.q[1] = *(const uint4*)(krow + kb + 16);
        sacc[tn] = __builtin_amdgcn_wmma_f32_16x16x32_bf16(
            false, qa[s].v, false, kf.v, (short)0, sacc[tn], false, false);
      }
    }

    // online softmax update per row (8 rows per lane group)
#pragma unroll
    for (int i = 0; i < 8; ++i) {
      float mloc = -3.0e38f;
#pragma unroll
      for (int tn = 0; tn < 4; ++tn) {
        float sv = sacc[tn][i] * scale;
        sacc[tn][i] = sv;
        mloc = fmaxf(mloc, sv);
      }
#pragma unroll
      for (int m = 1; m < 16; m <<= 1) mloc = fmaxf(mloc, __shfl_xor(mloc, m, 32));
      float mnew  = fmaxf(runmax[i], mloc);
      float alpha = __expf(runmax[i] - mnew);
      float psum = 0.0f;
#pragma unroll
      for (int tn = 0; tn < 4; ++tn) {
        float p = __expf(sacc[tn][i] - mnew);
        sacc[tn][i] = p;
        psum += p;
      }
#pragma unroll
      for (int m = 1; m < 16; m <<= 1) psum += __shfl_xor(psum, m, 32);
      Lsum[i] = Lsum[i] * alpha + psum;
      runmax[i] = mnew;
#pragma unroll
      for (int te = 0; te < 4; ++te) o[te][i] *= alpha;
#pragma unroll
      for (int tn = 0; tn < 4; ++tn)
        Pl[wv][rbase + i][tn * 16 + col] = f2bf(sacc[tn][i]);
    }
    __syncthreads();

    // O += P (16 x 64keys) . V (64keys x 64e)
#pragma unroll
    for (int s = 0; s < 2; ++s) {
      FragB pa;
      {
        const u16* prow = &Pl[wv][col][0];
        int kb = s * 32 + (half ? 8 : 0);
        pa.q[0] = *(const uint4*)(prow + kb);
        pa.q[1] = *(const uint4*)(prow + kb + 16);
      }
#pragma unroll
      for (int te = 0; te < 4; ++te) {
        FragB vf;
        const u16* vr = &Vt[te * 16 + col][0];
        int kb = s * 32 + (half ? 8 : 0);
        vf.q[0] = *(const uint4*)(vr + kb);
        vf.q[1] = *(const uint4*)(vr + kb + 16);
        o[te] = __builtin_amdgcn_wmma_f32_16x16x32_bf16(
            false, pa.v, false, vf.v, (short)0, o[te], false, false);
      }
    }
    __syncthreads();
  }

  // normalize and store to concat layout [B,S,H*DH] (bf16)
  float inv[8];
#pragma unroll
  for (int i = 0; i < 8; ++i) inv[i] = 1.0f / Lsum[i];
#pragma unroll
  for (int te = 0; te < 4; ++te) {
#pragma unroll
    for (int i = 0; i < 8; ++i) {
      int srow = q0 + rbase + i;
      int e = te * 16 + col;
      Ctx[((size_t)b * Sq + srow) * Dm + h * DHd + e] = f2bf(o[te][i] * inv[i]);
    }
  }
}

// ---------------------------------------------------------------------------
// LayerNorm (unbiased variance, ddof=1), one workgroup per row of 1024.
// Writes f32 out; optionally bf16 mirror for the next GEMM's A operand.
// ---------------------------------------------------------------------------
__global__ __launch_bounds__(256)
void layernorm_k(const float* __restrict__ in, const float* __restrict__ gam,
                 const float* __restrict__ bet,
                 float* __restrict__ outf, u16* __restrict__ outb) {
  __shared__ float rs[256], rq[256];
  const int row = blockIdx.x;
  const int tid = threadIdx.x;
  const int c = tid * 4;

  float4 xv = *(const float4*)(in + (size_t)row * Dm + c);
  float s = xv.x + xv.y + xv.z + xv.w;
  float q = xv.x * xv.x + xv.y * xv.y + xv.z * xv.z + xv.w * xv.w;
  rs[tid] = s; rq[tid] = q;
  __syncthreads();
  for (int st = 128; st > 0; st >>= 1) {
    if (tid < st) { rs[tid] += rs[tid + st]; rq[tid] += rq[tid + st]; }
    __syncthreads();
  }
  float mean = rs[0] * (1.0f / Dm);
  float var  = (rq[0] - (float)Dm * mean * mean) * (1.0f / (Dm - 1));
  float rstd = rsqrtf(var + 1e-5f);

  float4 gv = *(const float4*)(gam + c);
  float4 bv = *(const float4*)(bet + c);
  float4 ov;
  ov.x = (xv.x - mean) * rstd * gv.x + bv.x;
  ov.y = (xv.y - mean) * rstd * gv.y + bv.y;
  ov.z = (xv.z - mean) * rstd * gv.z + bv.z;
  ov.w = (xv.w - mean) * rstd * gv.w + bv.w;
  *(float4*)(outf + (size_t)row * Dm + c) = ov;
  if (outb) {
    u16* ob = outb + (size_t)row * Dm + c;
    ob[0] = f2bf(ov.x); ob[1] = f2bf(ov.y); ob[2] = f2bf(ov.z); ob[3] = f2bf(ov.w);
  }
}

// ---------------------------------------------------------------------------
// Host-side launcher
// ---------------------------------------------------------------------------
extern "C" void kernel_launch(void* const* d_in, const int* in_sizes, int n_in,
                              void* d_out, int out_size, void* d_ws, size_t ws_size,
                              hipStream_t stream) {
  (void)in_sizes; (void)n_in; (void)out_size; (void)ws_size;

  const float* x  = (const float*)d_in[0];
  const float* Wq = (const float*)d_in[1];
  const float* Wk = (const float*)d_in[2];
  const float* Wv = (const float*)d_in[3];
  const float* Wo = (const float*)d_in[4];
  const float* bo = (const float*)d_in[5];
  const float* g1 = (const float*)d_in[6];
  const float* b1 = (const float*)d_in[7];
  const float* W1 = (const float*)d_in[8];
  const float* c1 = (const float*)d_in[9];
  const float* W2 = (const float*)d_in[10];
  const float* c2 = (const float*)d_in[11];
  const float* g2 = (const float*)d_in[12];
  const float* b2 = (const float*)d_in[13];

  constexpr size_t NX  = (size_t)Mrows * Dm;       // 8M
  constexpr size_t NWH = (size_t)Hh * Dm * DHd;    // 1M
  constexpr size_t NWO = (size_t)Dm * Dm;          // 1M
  constexpr size_t NW1 = (size_t)Dm * DFF;         // 4M
  constexpr size_t NFF = (size_t)Mrows * DFF;      // 32M

  char* wp = (char*)d_ws;
  auto alloc = [&](size_t bytes) {
    char* p = wp;
    wp += (bytes + 255) & ~(size_t)255;
    return p;
  };
  u16*   xb  = (u16*)  alloc(NX * 2);
  u16*   wqb = (u16*)  alloc(NWH * 2);
  u16*   wkb = (u16*)  alloc(NWH * 2);
  u16*   wvb = (u16*)  alloc(NWH * 2);
  u16*   wob = (u16*)  alloc(NWO * 2);
  u16*   w1b = (u16*)  alloc(NW1 * 2);
  u16*   w2b = (u16*)  alloc(NW1 * 2);
  u16*   qb  = (u16*)  alloc(NX * 2);
  u16*   kb  = (u16*)  alloc(NX * 2);
  u16*   vb  = (u16*)  alloc(NX * 2);
  u16*   cc  = (u16*)  alloc(NX * 2);
  float* t1  = (float*)alloc(NX * 4);
  float* hf  = (float*)alloc(NX * 4);
  u16*   hb  = (u16*)  alloc(NX * 2);
  u16*   ff1 = (u16*)  alloc(NFF * 2);
  float* t2  = (float*)alloc(NX * 4);

  // 1) fp32 -> bf16 conversions
  cvt_bf16_k<<<2048, 256, 0, stream>>>(x,  xb,  NX);
  cvt_bf16_k<<<512,  256, 0, stream>>>(Wq, wqb, NWH);
  cvt_bf16_k<<<512,  256, 0, stream>>>(Wk, wkb, NWH);
  cvt_bf16_k<<<512,  256, 0, stream>>>(Wv, wvb, NWH);
  cvt_bf16_k<<<512,  256, 0, stream>>>(Wo, wob, NWO);
  cvt_bf16_k<<<1024, 256, 0, stream>>>(W1, w1b, NW1);
  cvt_bf16_k<<<1024, 256, 0, stream>>>(W2, w2b, NW1);

  // 2) QKV projections -> bf16 [B,H,S,DH]
  qkv_proj_k<<<dim3(Mrows / 256, Hh, 3), 256, 0, stream>>>(
      xb, wqb, wkb, wvb, qb, kb, vb);

  // 3) flash attention -> bf16 concat [B,S,D]
  flash_attn_k<<<dim3(Sq / 128, Hh, Bc), 256, 0, stream>>>(qb, kb, vb, cc);

  // 4) attn_out = concat @ Wo + bo + x  (f32)
  gemm_bf16_k<1><<<dim3(Dm / 64, Mrows / 256), 256, 0, stream>>>(
      cc, wob, bo, x, t1, nullptr, Mrows, Dm, Dm);

  // 5) LN1 -> h (f32 + bf16)
  layernorm_k<<<Mrows, 256, 0, stream>>>(t1, g1, b1, hf, hb);

  // 6) ff1 = gelu(h @ W1 + c1)  (bf16)
  gemm_bf16_k<2><<<dim3(DFF / 64, Mrows / 256), 256, 0, stream>>>(
      hb, w1b, c1, nullptr, nullptr, ff1, Mrows, DFF, Dm);

  // 7) t2 = ff1 @ W2 + c2 + h  (f32)
  gemm_bf16_k<1><<<dim3(Dm / 64, Mrows / 256), 256, 0, stream>>>(
      ff1, w2b, c2, hf, t2, nullptr, Mrows, Dm, DFF);

  // 8) LN2 -> d_out (f32)
  layernorm_k<<<Mrows, 256, 0, stream>>>(t2, g2, b2, (float*)d_out, nullptr);
}
